// OptRGLRU_50113678409895
// MI455X (gfx1250) — matrix operand order repaired and checked by reading, not compiled, not run on hardware
//
#include <hip/hip_runtime.h>

// RG-LRU for MI455X (gfx1250, wave32, WMMA + TDM).
// Shapes: B=4, S=4096, W=2048, H=16, BW=128.

typedef __attribute__((ext_vector_type(16))) _Float16     v16h;
typedef __attribute__((ext_vector_type(8)))  float        v8f;
typedef __attribute__((ext_vector_type(2)))  float        f32x2;
typedef __attribute__((ext_vector_type(4)))  unsigned int u32x4;
typedef __attribute__((ext_vector_type(8)))  int          i32x8;
typedef __attribute__((ext_vector_type(4)))  int          i32x4;

#define Bc   4
#define Sc   4096
#define Wc   2048
#define Hn   16
#define BWc  128
#define CS   256                 // sequence steps per chunk
#define NCHUNK (Sc / CS)         // 16

// ---------------------------------------------------------------------------
// TDM: async-load a 128x128 f16 tile (32 KB) from global into LDS.
// D# per CDNA5 ISA ch.8: group0 = {count=1, lds_addr, global_addr, type=2},
// group1 = {data_size=2B, tensor 128x128, tile 128x128, dim0_stride=128}.
// Groups 2/3 zero (<=2D tensor). Tracked by TENSORcnt.
// ---------------------------------------------------------------------------
__device__ __forceinline__ void tdm_load_tile_f16(const _Float16* gsrc, _Float16* ldst)
{
    const unsigned long long ga = (unsigned long long)(size_t)gsrc;
    const unsigned lds_off      = (unsigned)(size_t)ldst;   // low 32 bits = LDS offset

    u32x4 g0;
    g0[0] = 1u;                                              // count=1, user descriptor
    g0[1] = lds_off;                                         // lds_addr (bytes)
    g0[2] = (unsigned)(ga & 0xFFFFFFFFu);                    // global_addr[31:0]
    g0[3] = (unsigned)((ga >> 32) & 0x01FFFFFFu) | (2u << 30); // addr[56:32] | type=2

    i32x8 g1;
    g1[0] = (1 << 16);        // workgroup_mask=0, data_size=1 (2 bytes)
    g1[1] = (BWc << 16);      // tensor_dim0[15:0]   (bits 79:48)
    g1[2] = (BWc << 16);      // tensor_dim1[15:0]   (bits 111:80)
    g1[3] = (BWc << 16);      // tile_dim0           (bits 127:112)
    g1[4] = BWc;              // tile_dim1           (bits 143:128)
    g1[5] = BWc;              // tensor_dim0_stride  (bits 207:160, low 32)
    g1[6] = 0;
    g1[7] = 0;

    const i32x4 z4 = {0, 0, 0, 0};
#if defined(__clang_major__) && (__clang_major__ >= 23)
    const i32x8 z8 = {0, 0, 0, 0, 0, 0, 0, 0};
    __builtin_amdgcn_tensor_load_to_lds(g0, g1, z4, z4, z8, 0);
#else
    __builtin_amdgcn_tensor_load_to_lds(g0, g1, z4, z4, 0);
#endif
}

// ---------------------------------------------------------------------------
// K0: one-time f32 -> f16 conversion of the block-diagonal weights so the TDM
// can stream them straight into LDS in WMMA operand precision. 2 MB -> 1 MB.
// ---------------------------------------------------------------------------
__global__ __launch_bounds__(256) void rglru_cvt_weights(
    const float* __restrict__ ig_w, const float* __restrict__ ag_w,
    _Float16* __restrict__ wIGh, _Float16* __restrict__ wAGh)
{
    const int i = blockIdx.x * blockDim.x + threadIdx.x;   // over H*BW*BW
    if (i < Hn * BWc * BWc) {
        wIGh[i] = (_Float16)ig_w[i];
        wAGh[i] = (_Float16)ag_w[i];
    }
}

// ---------------------------------------------------------------------------
// K1: per (batch, head, chunk) block. 8 waves; wave w owns columns [16w,16w+16)
// of the head. TDM stages the head's f16 weights in LDS; both gate matmuls run
// as v_wmma_f32_16x16x32_f16; the RG-LRU elementwise math and the chunk-local
// scan (h_init = 0) happen in registers, writing:
//   y[b,s,w]    : chunk-local recurrence output (fixed up by K3)
//   acum[b,s,w] : cumulative product of a_eff from chunk start through s
//   sumA/sumH   : per-chunk summaries for the inter-chunk combine (K2)
// Resets (segment_pos==0) are encoded as a_eff = 0.
// ---------------------------------------------------------------------------
__global__ __launch_bounds__(256) void rglru_gates_scan(
    const float* __restrict__ x, const int* __restrict__ segpos,
    const _Float16* __restrict__ wIGh, const float* __restrict__ ig_b,
    const _Float16* __restrict__ wAGh, const float* __restrict__ ag_b,
    const float* __restrict__ a_param,
    float* __restrict__ y, float* __restrict__ acum,
    float* __restrict__ sumA, float* __restrict__ sumH)
{
    __shared__ _Float16 ldsIG[BWc * BWc];   // 32 KB
    __shared__ _Float16 ldsAG[BWc * BWc];   // 32 KB

    const int blk = blockIdx.x;
    const int c   = blk % NCHUNK;
    const int h   = (blk / NCHUNK) % Hn;
    const int b   = blk / (NCHUNK * Hn);

    const int tid   = threadIdx.x;
    const int lane  = tid & 31;
    const int wv    = tid >> 5;        // wave id 0..7 -> column slice
    const int nloc  = lane & 15;
    const int hi    = lane >> 4;       // 0: rows 0-7 / K-low half, 1: rows 8-15 / K-high half
    const int jbase = wv * 16;
    const int col   = jbase + nloc;          // column within head
    const int gcol  = h * BWc + col;         // column within W

    // Wave 0 kicks off the async weight staging; TENSORcnt is per-wave, so the
    // issuing wave drains it, then the workgroup barrier publishes the LDS.
    if (wv == 0) {
        tdm_load_tile_f16(wIGh + (size_t)h * BWc * BWc, ldsIG);
        tdm_load_tile_f16(wAGh + (size_t)h * BWc * BWc, ldsAG);
    }

    // Per-lane channel constants overlap with the TDM transfer.
    const float apv  = a_param[gcol];
    const float spv  = logf(1.0f + expf(apv));     // softplus(a_param)
    const float c8sp = -8.0f * spv;
    const float bIG  = ig_b[h * BWc + col];
    const float bAG  = ag_b[h * BWc + col];

    if (wv == 0) __builtin_amdgcn_s_wait_tensorcnt(0);
    __syncthreads();

    // B fragments (weights), loaded once into VGPRs.
    // B 32x16 in the contiguous k order used here (same as A): VGPR v holds K = 2v,2v+1 (+16 on upper lanes), N = lane&15.
    v16h bIGf[4], bAGf[4];
#pragma unroll
    for (int kt = 0; kt < 4; ++kt) {
        const int kb = kt * 32 + hi * 16;
        v16h t0, t1;
#pragma unroll
        for (int v = 0; v < 8; ++v) {
            const int k = kb + 2 * v;
            t0[2*v]   = ldsIG[k * BWc + col];
            t0[2*v+1] = ldsIG[(k + 1) * BWc + col];
            t1[2*v]   = ldsAG[k * BWc + col];
            t1[2*v+1] = ldsAG[(k + 1) * BWc + col];
        }
        bIGf[kt] = t0; bAGf[kt] = t1;
    }

    float hcarry = 0.0f;   // chunk-local h entering current subtile (all lanes)
    float aprev  = 1.0f;   // cumulative product from chunk start through prev subtile

    const int    rowChunk0 = c * CS;
    const float* xb        = x + (size_t)b * Sc * Wc;

    for (int st = 0; st < CS / 16; ++st) {
        const int row0 = rowChunk0 + st * 16;

        // A fragments: A 16x32 in contiguous k order (same as B). Lane holds row M = lane&15;
        // VGPR v pairs K = 2v,2v+1 (+16 on upper lanes). Pairs are
        // contiguous in memory -> float2 load + cvt (merges to b128 clauses).
        const int    arow = row0 + nloc;
        const float* xrow = xb + (size_t)arow * Wc + h * BWc;
        if (st + 1 < CS / 16)
            __builtin_prefetch(xrow + (size_t)16 * Wc, 0, 0);  // global_prefetch_b8
        v16h af[4];
#pragma unroll
        for (int kt = 0; kt < 4; ++kt) {
            v16h t;
#pragma unroll
            for (int v = 0; v < 8; ++v) {
                const int kk = kt * 32 + hi * 16 + 2 * v;
                const f32x2 p = *(const f32x2*)(xrow + kk);
                t[2*v]   = (_Float16)p.x;
                t[2*v+1] = (_Float16)p.y;
            }
            af[kt] = t;
        }

        v8f accI = {}; v8f accA = {};
#pragma unroll
        for (int kt = 0; kt < 4; ++kt) {
            accI = __builtin_amdgcn_wmma_f32_16x16x32_f16(
                false, af[kt], false, bIGf[kt], (short)0, accI, false, false);
            accA = __builtin_amdgcn_wmma_f32_16x16x32_f16(
                false, af[kt], false, bAGf[kt], (short)0, accA, false, false);
        }

        // Elementwise gates + local scan over this lane's 8 rows (h_init = 0).
        float yloc[8], cum[8];
        float hl = 0.0f, pl = 1.0f;
#pragma unroll
        for (int r = 0; r < 8; ++r) {
            const int   row   = row0 + hi * 8 + r;
            const float lx    = accI[r] + bIG;
            const float la    = accA[r] + bAG;
            const float gx    = 1.0f / (1.0f + expf(-lx));
            const float ga    = 1.0f / (1.0f + expf(-la));
            const float log_a = c8sp * ga;
            const float a     = expf(log_a);
            const bool  reset = (segpos[b * Sc + row] == 0);
            const float mult  = reset ? 1.0f
                                      : sqrtf(fmaxf(0.0f, 1.0f - expf(2.0f * log_a)));
            const float xv    = xb[(size_t)row * Wc + gcol];
            const float xn    = xv * gx * mult;
            const float aeff  = reset ? 0.0f : a;
            hl = aeff * hl + xn;
            pl *= aeff;
            yloc[r] = hl;
            cum[r]  = pl;
        }

        // Combine halves: rows 8-15 (upper lanes) need true h after row 7.
        const float t_low  = yloc[7] + cum[7] * hcarry;        // valid on low lanes
        const float hin_up = __shfl(t_low, nloc, 32);
        const float hin    = hi ? hin_up : hcarry;
        const float c7low  = __shfl(cum[7], nloc, 32);
        const float cmul   = hi ? c7low : 1.0f;

        float y15 = 0.0f, cfull7 = 1.0f;
#pragma unroll
        for (int r = 0; r < 8; ++r) {
            const int    row   = row0 + hi * 8 + r;
            const float  yt    = yloc[r] + cum[r] * hin;
            const float  cfull = cmul * cum[r];
            const size_t idx   = (size_t)b * Sc * Wc + (size_t)row * Wc + gcol;
            y[idx]    = yt;
            acum[idx] = aprev * cfull;
            if (r == 7) { y15 = yt; cfull7 = cfull; }
        }
        hcarry = __shfl(y15,    nloc + 16, 32);   // row-15 true value (upper lanes)
        aprev  = aprev * __shfl(cfull7, nloc + 16, 32);
    }

    if (hi == 0) {
        sumA[(size_t)c * Bc * Wc + b * Wc + gcol] = aprev;   // prod a over chunk
        sumH[(size_t)c * Bc * Wc + b * Wc + gcol] = hcarry;  // chunk-end h (init 0)
    }
}

// ---------------------------------------------------------------------------
// K2: inter-chunk sequential combine (per channel). NCHUNK=16 iterations.
// carryIn[c] = true h entering chunk c; also writes last_h output.
// ---------------------------------------------------------------------------
__global__ void rglru_carry(const float* __restrict__ cache,
                            const float* __restrict__ sumA,
                            const float* __restrict__ sumH,
                            float* __restrict__ carryIn,
                            float* __restrict__ lasth)
{
    const int i = blockIdx.x * blockDim.x + threadIdx.x;  // over B*W
    if (i >= Bc * Wc) return;
    float carry = cache[i];
#pragma unroll
    for (int cc = 0; cc < NCHUNK; ++cc) {
        carryIn[cc * Bc * Wc + i] = carry;
        carry = sumH[cc * Bc * Wc + i] + sumA[cc * Bc * Wc + i] * carry;
    }
    lasth[i] = carry;
}

// ---------------------------------------------------------------------------
// K3: fix-up pass: y_t += Acum_t * carryIn[chunk(t)]. float4 streams.
// ---------------------------------------------------------------------------
__global__ __launch_bounds__(256) void rglru_fix(float* __restrict__ y,
                                                 const float* __restrict__ acum,
                                                 const float* __restrict__ carryIn)
{
    const size_t i = ((size_t)blockIdx.x * blockDim.x + threadIdx.x) * 4;
    if (i >= (size_t)Bc * Sc * Wc) return;
    const int    w  = (int)(i % Wc);
    const size_t bs = i / Wc;
    const int    s  = (int)(bs % Sc);
    const int    b  = (int)(bs / Sc);
    const int    cc = s / CS;

    const float4 cr = *(const float4*)(carryIn + (size_t)cc * Bc * Wc + (size_t)b * Wc + w);
    const float4 av = *(const float4*)(acum + i);
    float4       yv = *(const float4*)(y + i);
    yv.x += av.x * cr.x;
    yv.y += av.y * cr.y;
    yv.z += av.z * cr.z;
    yv.w += av.w * cr.w;
    *(float4*)(y + i) = yv;
}

// ---------------------------------------------------------------------------
extern "C" void kernel_launch(void* const* d_in, const int* in_sizes, int n_in,
                              void* d_out, int out_size, void* d_ws, size_t ws_size,
                              hipStream_t stream)
{
    (void)in_sizes; (void)n_in; (void)out_size; (void)ws_size;

    const float* x       = (const float*)d_in[0];
    const int*   segpos  = (const int*)  d_in[1];
    const float* ig_w    = (const float*)d_in[2];
    const float* ig_b    = (const float*)d_in[3];
    const float* ag_w    = (const float*)d_in[4];
    const float* ag_b    = (const float*)d_in[5];
    const float* a_param = (const float*)d_in[6];
    const float* cache   = (const float*)d_in[7];

    float* y     = (float*)d_out;
    float* lasth = y + (size_t)Bc * Sc * Wc;

    // Workspace: Acum (B*S*W f32) | sumA | sumH | carryIn | f16 weights (ig, ag)
    float*     acum    = (float*)d_ws;
    float*     sA      = acum + (size_t)Bc * Sc * Wc;
    float*     sH      = sA + (size_t)NCHUNK * Bc * Wc;
    float*     carryIn = sH + (size_t)NCHUNK * Bc * Wc;
    _Float16*  wIGh    = (_Float16*)(carryIn + (size_t)NCHUNK * Bc * Wc);
    _Float16*  wAGh    = wIGh + (size_t)Hn * BWc * BWc;

    rglru_cvt_weights<<<(Hn * BWc * BWc + 255) / 256, 256, 0, stream>>>(
        ig_w, ag_w, wIGh, wAGh);

    rglru_gates_scan<<<Bc * Hn * NCHUNK, 256, 0, stream>>>(
        x, segpos, wIGh, ig_b, wAGh, ag_b, a_param, y, acum, sA, sH);

    rglru_carry<<<(Bc * Wc + 255) / 256, 256, 0, stream>>>(
        cache, sA, sH, carryIn, lasth);

    rglru_fix<<<(int)(((size_t)Bc * Sc * Wc / 4 + 255) / 256), 256, 0, stream>>>(
        y, acum, carryIn);
}
